// LongAttention_68272800137364
// MI455X (gfx1250) — matrix operand order repaired and checked by reading, not compiled, not run on hardware
//
#include <hip/hip_runtime.h>
#include <hip/hip_bf16.h>

// Problem constants (from reference)
#define BSZ 4
#define TLEN 2048
#define DMODEL 1024
#define HDDIM 2048
#define NHEAD 16
#define HDIM 128
#define NSEG 16
#define TSEG (TLEN / NSEG)

typedef __attribute__((ext_vector_type(16))) __bf16 v16bf;
typedef __attribute__((ext_vector_type(8)))  __bf16 v8bf;
typedef __attribute__((ext_vector_type(8)))  float  v8f;

// float -> bf16 (round-to-nearest-even) without relying on __bf16 cast support
__device__ __forceinline__ __bf16 f2bf(float f) {
    union { float f; unsigned u; } uf; uf.f = f;
    unsigned r = uf.u + 0x7FFFu + ((uf.u >> 16) & 1u);
    unsigned short h = (unsigned short)(r >> 16);
    union { unsigned short s; __bf16 b; } ub; ub.s = h;
    return ub.b;
}

// ---------------------------------------------------------------------------
// 1) Causal depthwise conv1d (kernel=4, left pad 3), writes f32 + bf16 copies
// ---------------------------------------------------------------------------
__global__ __launch_bounds__(256) void conv_kernel(
    const float* __restrict__ x, const float* __restrict__ w,
    const float* __restrict__ bias, float* __restrict__ xc,
    __bf16* __restrict__ xcb, int total)
{
    int idx = blockIdx.x * blockDim.x + threadIdx.x;
    if (idx >= total) return;
    int c = idx & (DMODEL - 1);
    int t = (idx >> 10) & (TLEN - 1);
    float acc = bias[c];
    const float* wr = w + c * 4;
#pragma unroll
    for (int j = 0; j < 4; ++j) {
        int tt = t - 3 + j;
        if (tt >= 0) acc += x[idx + (tt - t) * DMODEL] * wr[j];
    }
    xc[idx] = acc;
    xcb[idx] = f2bf(acc);
}

// ---------------------------------------------------------------------------
// 2) f32 -> bf16 convert (weights)
// ---------------------------------------------------------------------------
__global__ __launch_bounds__(256) void cvt_bf16_kernel(
    const float* __restrict__ in, __bf16* __restrict__ out, int n)
{
    int i = blockIdx.x * blockDim.x + threadIdx.x;
    if (i < n) out[i] = f2bf(in[i]);
}

// ---------------------------------------------------------------------------
// 3) bf16 WMMA GEMM:  C[M,N] = A[M,K] * W[N,K]^T  (f32 accumulate)
//    Wave tile 32x64 (2 M-tiles x 4 N-tiles), block = 4 waves along N.
//    Grid: (M/32, N/256).
// ---------------------------------------------------------------------------
__global__ __launch_bounds__(128) void gemm_bf16_wmma(
    const __bf16* __restrict__ A, const __bf16* __restrict__ W,
    float* __restrict__ C, int M, int N, int K)
{
    const int lane = threadIdx.x & 31;
    const int wave = threadIdx.x >> 5;
    const int l16  = lane & 15;
    const int half = lane >> 4;

    const int m0 = blockIdx.x * 32;
    const int n0 = blockIdx.y * 256 + wave * 64;

    v8f acc[2][4] = {};   // zero-init accumulators

    for (int k0 = 0; k0 < K; k0 += 32) {
        v16bf a[2], b[4];
        // A fragment: lane row = m, half selects K 0-15 / 16-31 (contiguous)
#pragma unroll
        for (int mt = 0; mt < 2; ++mt) {
            const __bf16* ap = A + (size_t)(m0 + mt * 16 + l16) * K + k0 + half * 16;
            v8bf lo = *(const v8bf*)ap;          // K = k0 + half*16     .. +7
            v8bf hi = *(const v8bf*)(ap + 8);    // K = k0 + half*16 + 8 .. +7
#pragma unroll
            for (int e = 0; e < 8; ++e) { a[mt][e] = lo[e]; a[mt][8 + e] = hi[e]; }
        }
        // B fragment: lane col = n, half selects K 0-15 / 16-31 (contiguous in W row)
#pragma unroll
        for (int nt = 0; nt < 4; ++nt) {
            const __bf16* bp = W + (size_t)(n0 + nt * 16 + l16) * K + k0 + half * 16;
            b[nt] = *(const v16bf*)bp;
        }
#pragma unroll
        for (int mt = 0; mt < 2; ++mt)
#pragma unroll
            for (int nt = 0; nt < 4; ++nt)
                acc[mt][nt] = __builtin_amdgcn_wmma_f32_16x16x32_bf16(
                    false, a[mt], false, b[nt], (short)0, acc[mt][nt], false, false);
    }

    // D layout: VGPR r -> row m = r + 8*half, col n = l16
#pragma unroll
    for (int mt = 0; mt < 2; ++mt) {
#pragma unroll
        for (int nt = 0; nt < 4; ++nt) {
            const int col = n0 + nt * 16 + l16;
#pragma unroll
            for (int r = 0; r < 8; ++r) {
                const int row = m0 + mt * 16 + half * 8 + r;
                C[(size_t)row * N + col] = acc[mt][nt][r];
            }
        }
    }
}

// ---------------------------------------------------------------------------
// 4) In-place L2 normalize of qk over D=128 per (b,t,h)
// ---------------------------------------------------------------------------
__global__ __launch_bounds__(256) void normalize_kernel(float* qk, int ngroups)
{
    int gid = blockIdx.x * blockDim.x + threadIdx.x;
    if (gid >= ngroups) return;
    float* p = qk + (size_t)gid * HDIM;
    float ss = 0.f;
#pragma unroll
    for (int i = 0; i < HDIM / 4; ++i) {
        float4 v = ((const float4*)p)[i];
        ss += v.x * v.x + v.y * v.y + v.z * v.z + v.w * v.w;
    }
    float inv = 1.f / fmaxf(sqrtf(ss), 1e-12f);
#pragma unroll
    for (int i = 0; i < HDIM / 4; ++i) {
        float4 v = ((float4*)p)[i];
        v.x *= inv; v.y *= inv; v.z *= inv; v.w *= inv;
        ((float4*)p)[i] = v;
    }
}

// ---------------------------------------------------------------------------
// 5) state_r * decay, written in place over the v buffer
//    sc = (v*cos(t*f) + v*q_{t-1}) * sigmoid(g_raw + g_b) * gamma^t
// ---------------------------------------------------------------------------
__global__ __launch_bounds__(256) void state_kernel(
    const float* __restrict__ q, float* vbuf,
    const float* __restrict__ graw, const float* __restrict__ g_b,
    const float* __restrict__ freqs, const float* __restrict__ gamma, int total)
{
    int idx = blockIdx.x * blockDim.x + threadIdx.x;
    if (idx >= total) return;
    int d = idx & (HDIM - 1);
    int h = (idx >> 7) & (NHEAD - 1);
    int t = (idx >> 11) & (TLEN - 1);
    int e = (h << 7) + d;
    float vv = vbuf[idx];
    float qprev = (t > 0) ? q[idx - HDDIM] : 0.f;
    float ang = (float)t * freqs[e];
    float gg = 1.f / (1.f + __expf(-(graw[idx] + g_b[e])));
    float dec = __powf(gamma[h], (float)t);
    vbuf[idx] = (vv * __cosf(ang) + vv * qprev) * gg * dec;
}

// ---------------------------------------------------------------------------
// 6) Segment sums for the T-scan  (grid: (B*H, NSEG), block: D=128)
// ---------------------------------------------------------------------------
__global__ __launch_bounds__(128) void seg_sum_kernel(
    const float* __restrict__ sc, float* __restrict__ S)
{
    int bh = blockIdx.x, seg = blockIdx.y, d = threadIdx.x;
    int b = bh >> 4, h = bh & (NHEAD - 1);
    size_t base = (((size_t)b * TLEN) * NHEAD + h) * HDIM + d;
    float s = 0.f;
    int t0 = seg * TSEG;
    for (int i = 0; i < TSEG; ++i)
        s += sc[base + (size_t)(t0 + i) * HDDIM];
    S[((size_t)bh * NSEG + seg) * HDIM + d] = s;
}

// ---------------------------------------------------------------------------
// 7) Segmented scan: mem = cumsum(sc)/(gamma^t + 1e-6); out_e = mem*q  (bf16)
// ---------------------------------------------------------------------------
__global__ __launch_bounds__(128) void seg_scan_kernel(
    const float* __restrict__ sc, const float* __restrict__ q,
    const float* __restrict__ S, const float* __restrict__ gamma,
    __bf16* __restrict__ oute)
{
    int bh = blockIdx.x, seg = blockIdx.y, d = threadIdx.x;
    int b = bh >> 4, h = bh & (NHEAD - 1);
    float acc = 0.f;
    for (int j = 0; j < seg; ++j)
        acc += S[((size_t)bh * NSEG + j) * HDIM + d];
    float gm = gamma[h];
    size_t base = (((size_t)b * TLEN) * NHEAD + h) * HDIM + d;
    int t0 = seg * TSEG;
    for (int i = 0; i < TSEG; ++i) {
        int t = t0 + i;
        size_t idx = base + (size_t)t * HDDIM;
        acc += sc[idx];
        float dec = __powf(gm, (float)t);
        float mem = acc / (dec + 1e-6f);
        oute[idx] = f2bf(mem * q[idx]);   // [B,T,H*D] flat == same index
    }
}

// ---------------------------------------------------------------------------
// 9) LayerNorm over DMODEL per row
// ---------------------------------------------------------------------------
__global__ __launch_bounds__(256) void layernorm_kernel(
    const float* __restrict__ y, const float* __restrict__ lng,
    const float* __restrict__ lnb, float* __restrict__ out)
{
    int row = blockIdx.x;
    const float* p = y + (size_t)row * DMODEL;
    float s = 0.f, ss = 0.f;
    for (int i = threadIdx.x; i < DMODEL; i += 256) {
        float v = p[i]; s += v; ss += v * v;
    }
#pragma unroll
    for (int off = 16; off > 0; off >>= 1) {
        s  += __shfl_down(s,  off, 32);
        ss += __shfl_down(ss, off, 32);
    }
    __shared__ float sh_s[8], sh_ss[8];
    int wv = threadIdx.x >> 5, ln = threadIdx.x & 31;
    if (ln == 0) { sh_s[wv] = s; sh_ss[wv] = ss; }
    __syncthreads();
    if (threadIdx.x == 0) {
        float ts = 0.f, tss = 0.f;
        for (int w = 0; w < 8; ++w) { ts += sh_s[w]; tss += sh_ss[w]; }
        sh_s[0] = ts; sh_ss[0] = tss;
    }
    __syncthreads();
    float mu  = sh_s[0] * (1.f / DMODEL);
    float var = sh_ss[0] * (1.f / DMODEL) - mu * mu;
    float inv = rsqrtf(var + 1e-5f);
    float* op = out + (size_t)row * DMODEL;
    for (int i = threadIdx.x; i < DMODEL; i += 256)
        op[i] = (p[i] - mu) * inv * lng[i] + lnb[i];
}

// ---------------------------------------------------------------------------
extern "C" void kernel_launch(void* const* d_in, const int* in_sizes, int n_in,
                              void* d_out, int out_size, void* d_ws, size_t ws_size,
                              hipStream_t stream)
{
    (void)in_sizes; (void)n_in; (void)out_size; (void)ws_size;
    const float* x      = (const float*)d_in[0];
    const float* conv_w = (const float*)d_in[1];
    const float* conv_b = (const float*)d_in[2];
    const float* qk_w   = (const float*)d_in[3];
    const float* v_w    = (const float*)d_in[4];
    const float* g_w    = (const float*)d_in[5];
    const float* g_b    = (const float*)d_in[6];
    const float* o_w    = (const float*)d_in[7];
    const float* ln_g   = (const float*)d_in[8];
    const float* ln_b   = (const float*)d_in[9];
    const float* freqs  = (const float*)d_in[10];
    const float* gamma  = (const float*)d_in[11];
    float* out = (float*)d_out;

    const size_t M = (size_t)BSZ * TLEN;          // 8192 rows
    char* ws = (char*)d_ws;
    size_t off = 0;
    auto alloc = [&](size_t bytes) -> void* {
        void* p = ws + off; off += (bytes + 255) & ~(size_t)255; return p;
    };
    float*  xc    = (float*) alloc(M * DMODEL * 4);
    __bf16* xcb   = (__bf16*)alloc(M * DMODEL * 2);
    __bf16* qkwb  = (__bf16*)alloc((size_t)HDDIM * DMODEL * 2);
    __bf16* vwb   = (__bf16*)alloc((size_t)HDDIM * DMODEL * 2);
    __bf16* gwb   = (__bf16*)alloc((size_t)HDDIM * DMODEL * 2);
    __bf16* owb   = (__bf16*)alloc((size_t)DMODEL * HDDIM * 2);
    float*  qbuf  = (float*) alloc(M * HDDIM * 4);   // qk -> q (in place)
    float*  vbuf  = (float*) alloc(M * HDDIM * 4);   // v  -> sc (in place)
    float*  gbuf  = (float*) alloc(M * HDDIM * 4);
    __bf16* oute  = (__bf16*)alloc(M * HDDIM * 2);
    float*  Sbuf  = (float*) alloc((size_t)BSZ * NHEAD * NSEG * HDIM * 4);
    float*  preln = (float*) alloc(M * DMODEL * 4);

    // 1) conv + bf16 activations
    {
        int total = (int)(M * DMODEL);
        conv_kernel<<<total / 256, 256, 0, stream>>>(x, conv_w, conv_b, xc, xcb, total);
    }
    // 2) weight converts
    {
        int nw = HDDIM * DMODEL;
        cvt_bf16_kernel<<<nw / 256, 256, 0, stream>>>(qk_w, qkwb, nw);
        cvt_bf16_kernel<<<nw / 256, 256, 0, stream>>>(v_w,  vwb,  nw);
        cvt_bf16_kernel<<<nw / 256, 256, 0, stream>>>(g_w,  gwb,  nw);
        cvt_bf16_kernel<<<nw / 256, 256, 0, stream>>>(o_w,  owb,  nw);
    }
    // 3) projections: [8192,1024] x [2048,1024]^T
    {
        dim3 grid((unsigned)(M / 32), HDDIM / 256);
        gemm_bf16_wmma<<<grid, 128, 0, stream>>>(xcb, qkwb, qbuf, (int)M, HDDIM, DMODEL);
        gemm_bf16_wmma<<<grid, 128, 0, stream>>>(xcb, vwb,  vbuf, (int)M, HDDIM, DMODEL);
        gemm_bf16_wmma<<<grid, 128, 0, stream>>>(xcb, gwb,  gbuf, (int)M, HDDIM, DMODEL);
    }
    // 4) normalize qk -> q (in place)
    {
        int ngroups = (int)(M * NHEAD);
        normalize_kernel<<<ngroups / 256, 256, 0, stream>>>(qbuf, ngroups);
    }
    // 5) state * decay -> sc (in place over vbuf)
    {
        int total = (int)(M * HDDIM);
        state_kernel<<<total / 256, 256, 0, stream>>>(qbuf, vbuf, gbuf, g_b, freqs, gamma, total);
    }
    // 6/7) segmented scan along T
    {
        dim3 grid(BSZ * NHEAD, NSEG);
        seg_sum_kernel <<<grid, HDIM, 0, stream>>>(vbuf, Sbuf);
        seg_scan_kernel<<<grid, HDIM, 0, stream>>>(vbuf, qbuf, Sbuf, gamma, oute);
    }
    // 8) output projection: [8192,2048] x [1024,2048]^T
    {
        dim3 grid((unsigned)(M / 32), DMODEL / 256);
        gemm_bf16_wmma<<<grid, 128, 0, stream>>>(oute, owb, preln, (int)M, DMODEL, HDDIM);
    }
    // 9) LayerNorm -> d_out (f32)
    layernorm_kernel<<<(unsigned)M, 256, 0, stream>>>(preln, ln_g, ln_b, out);
}
